// SpectrogramExtractor_67826123538945
// MI455X (gfx1250) — matrix operand order repaired and checked by reading, not compiled, not run on hardware
//
#include <hip/hip_runtime.h>
#include <cmath>

typedef __attribute__((ext_vector_type(16))) _Float16 v16h;
typedef __attribute__((ext_vector_type(8)))  _Float16 v8h;
typedef __attribute__((ext_vector_type(8)))  float    v8f;
typedef __attribute__((ext_vector_type(4)))  unsigned int v4u;
typedef __attribute__((ext_vector_type(8)))  int      v8i;
typedef __attribute__((ext_vector_type(4)))  int      v4i;

#define BATCH 8
#define T_LEN 882000
#define HOP   441
#define N0    2002      // ceil((882000 + 512)/441)
#define NMEL  80

// One workgroup = 32 frames (2 tiles/wave) of one batch, all bins of one FFT size.
// 8 waves (wave32). Spectral DFT and mel projection via v_wmma_f32_16x16x32_f16.
// Mel tiles are DMA'd global->LDS with the Tensor Data Mover, overlapped with
// the spectral K-loop, published with s_wait_tensorcnt + barrier.
template<int F>
__global__ __launch_bounds__(256)
void spectro_wmma(const float* __restrict__ x,
                  const float* __restrict__ window,
                  const float* __restrict__ mel,
                  float* __restrict__ out, int fi)
{
  constexpr int H     = F / 2;
  constexpr int BINS  = H + 1;
  constexpr int PB    = 128;                      // bins per pass (8 waves x 16)
  constexpr int NPASS = (BINS + PB - 1) / PB;     // 5 / 9 / 17
  constexpr int NCH   = F / 32;                   // K-chunks per pass
  constexpr int LDA   = F + 8;                    // halves (bank-stagger pad)
  constexpr int SLD   = PB + 8;                   // 136 halves

  __shared__ _Float16 Alds[32 * LDA];             // windowed, half-swapped frames (f16)
  __shared__ _Float16 stage[32 * SLD];            // power-spectrum tiles (f16)
  __shared__ float    mlds[PB * NMEL];            // mel chunk (f32), [k][j], TDM target

  const int b    = blockIdx.y;
  const int fr0  = blockIdx.x * 32;
  const int tid  = threadIdx.x;
  const int wave = tid >> 5;
  const int lane = tid & 31;
  const int lm   = lane & 15;
  const int lh   = lane >> 4;                     // 0 | 1 (lane half)

  const float* xb = x + (size_t)b * T_LEN;

  // ---- stage frames: front-zero-padded tail frames, window, fftshift ----
  for (int idx = tid; idx < 32 * F; idx += 256) {
    const int m = idx / F;
    const int j = idx - m * F;                    // position after half-swap
    const int n = fr0 + m;
    const int i = (j < H) ? (j + H) : (j - H);    // original in-frame index
    int pad = n * HOP + F - (T_LEN + H);
    pad = pad > 0 ? pad : 0;
    float v = 0.0f;
    if (i >= pad) {
      const int t = n * HOP + i - pad;            // index into zero-front-padded wp
      if (t >= H) v = xb[t - H];
    }
    v *= window[i];
    Alds[m * LDA + j] = (_Float16)v;
  }

  v8f accMel0 = {}, accMel1 = {};                 // [frames x melcols], acc across passes

  for (int pass = 0; pass < NPASS; ++pass) {
    __syncthreads();   // Alds ready (pass 0); stage/mlds consumed (later passes)

    // ---- TDM: async-load mel chunk [PB x NMEL] f32 -> LDS (1D contiguous tile).
    // tensor_dim0 = remaining valid elements, so OOB-zero gives tail padding free.
    if (wave == 0) {
      const unsigned elems    = (unsigned)(PB * NMEL);          // 10240 tile elements
      const unsigned valid    = (unsigned)((BINS - pass * PB) > PB
                                   ? elems : (BINS - pass * PB) * NMEL);
      const unsigned long gaddr = (unsigned long)(mel + (size_t)pass * PB * NMEL);
      const unsigned ldsaddr  = (unsigned)(unsigned long)(uintptr_t)&mlds[0];
      v4u g0; v8i g1; v4i g2, g3; v8i g4;
      g0[0] = 1u;                                  // count=1 valid descriptor
      g0[1] = ldsaddr;                             // lds_addr
      g0[2] = (unsigned)(gaddr & 0xFFFFFFFFu);     // global_addr[31:0]
      g0[3] = (unsigned)((gaddr >> 32) & 0x1FFFFFFu) | 0x80000000u; // addr[56:32] | type=2
      g1[0] = 0x00020000;                          // data_size=2 (4B), no multicast/pad
      g1[1] = (int)((valid & 0xFFFFu) << 16);      // tensor_dim0[15:0]
      g1[2] = (int)((valid >> 16) | (1u << 16));   // tensor_dim0[31:16], tensor_dim1=1
      g1[3] = (int)(elems << 16);                  // tile_dim0 = 10240
      g1[4] = 0;                                   // tile_dim1=0 (1D), tile_dim2=0
      g1[5] = (int)elems;                          // tensor_dim0_stride (don't care, 1D)
      g1[6] = 0;
      g1[7] = 0;
      g2[0] = g2[1] = g2[2] = g2[3] = 0;
      g3[0] = g3[1] = g3[2] = g3[3] = 0;
      g4[0] = g4[1] = g4[2] = g4[3] = 0;
      g4[4] = g4[5] = g4[6] = g4[7] = 0;
      __builtin_amdgcn_tensor_load_to_lds(g0, g1, g2, g3, g4, 0);
    }

    // per-lane DFT column tables: cos/sin(2*pi*n*K/F), K = (e&7) + lh*8 + (e>>3)*16 (B layout)
    const int n = pass * PB + wave * 16 + lm;     // this lane's bin column
    float cd[16], sd[16];
    #pragma unroll
    for (int e = 0; e < 16; ++e) {
      const int K = (e & 7) + lh * 8 + (e >> 3) * 16;
      const int r = (n * K) & (F - 1);            // exact mod-F reduction
      sincosf(6.283185307179586f * ((float)r / (float)F), &sd[e], &cd[e]);
    }
    float cstep, sstep;
    {
      const int r = (n * 32) & (F - 1);
      sincosf(6.283185307179586f * ((float)r / (float)F), &sstep, &cstep);
    }
    float c0 = 1.0f, s0 = 0.0f;                   // rotation for k0 = 32*ch

    v8f accRe0 = {}, accIm0 = {}, accRe1 = {}, accIm1 = {};
    const _Float16* arow0 = &Alds[lm * LDA];
    const _Float16* arow1 = &Alds[(16 + lm) * LDA];
    for (int ch = 0; ch < NCH; ++ch) {
      const int kb = ch * 32 + lh * 8;
      // A fragments for both frame tiles (same K chunk)
      const v8h lo0 = *(const v8h*)(arow0 + kb);
      const v8h hi0 = *(const v8h*)(arow0 + kb + 16);
      const v8h lo1 = *(const v8h*)(arow1 + kb);
      const v8h hi1 = *(const v8h*)(arow1 + kb + 16);
      v16h a0, a1;
      #pragma unroll
      for (int e = 0; e < 8; ++e) {
        a0[e] = lo0[e]; a0[e + 8] = hi0[e];
        a1[e] = lo1[e]; a1[e + 8] = hi1[e];
      }

      // B fragments by angle addition: cos/sin(theta_n*(32*ch + K)); shared by both tiles
      v16h bre, bim;
      #pragma unroll
      for (int e = 0; e < 16; ++e) {
        bre[e] = (_Float16)(c0 * cd[e] - s0 * sd[e]);
        bim[e] = (_Float16)(s0 * cd[e] + c0 * sd[e]);
      }
      accRe0 = __builtin_amdgcn_wmma_f32_16x16x32_f16(false, a0, false, bre, (short)0, accRe0, false, false);
      accIm0 = __builtin_amdgcn_wmma_f32_16x16x32_f16(false, a0, false, bim, (short)0, accIm0, false, false);
      accRe1 = __builtin_amdgcn_wmma_f32_16x16x32_f16(false, a1, false, bre, (short)0, accRe1, false, false);
      accIm1 = __builtin_amdgcn_wmma_f32_16x16x32_f16(false, a1, false, bim, (short)0, accIm1, false, false);

      const float c1 = c0 * cstep - s0 * sstep;   // advance rotation by 32*theta_n
      s0 = s0 * cstep + c0 * sstep;
      c0 = c1;
    }

    // power spectrum tiles -> stage (f16), masked beyond BINS
    const float msk = (n < BINS) ? 1.0f : 0.0f;
    #pragma unroll
    for (int r = 0; r < 8; ++r) {
      const float p0 = (accRe0[r] * accRe0[r] + accIm0[r] * accIm0[r]) * msk;
      const float p1 = (accRe1[r] * accRe1[r] + accIm1[r] * accIm1[r]) * msk;
      stage[(r + lh * 8) * SLD + wave * 16 + lm]        = (_Float16)p0;
      stage[(16 + r + lh * 8) * SLD + wave * 16 + lm]   = (_Float16)p1;
    }
    __builtin_amdgcn_s_wait_tensorcnt(0);         // mel DMA done (no-op for waves 1..7)
    __syncthreads();

    // mel projection: stage(32x128) @ mlds(128x80) via WMMA (waves 0..4 own 16 cols)
    if (wave < 5) {
      const _Float16* srow0 = &stage[lm * SLD];
      const _Float16* srow1 = &stage[(16 + lm) * SLD];
      const int j = wave * 16 + lm;               // mel column, < 80
      #pragma unroll
      for (int ch = 0; ch < 4; ++ch) {
        const int kb = ch * 32 + lh * 8;
        v16h bm;
        #pragma unroll
        for (int e = 0; e < 16; ++e) {
          const int k = ch * 32 + (e & 7) + lh * 8 + (e >> 3) * 16;
          bm[e] = (_Float16)mlds[k * NMEL + j];
        }
        const v8h lo0 = *(const v8h*)(srow0 + kb);
        const v8h hi0 = *(const v8h*)(srow0 + kb + 16);
        const v8h lo1 = *(const v8h*)(srow1 + kb);
        const v8h hi1 = *(const v8h*)(srow1 + kb + 16);
        v16h a0, a1;
        #pragma unroll
        for (int e = 0; e < 8; ++e) {
          a0[e] = lo0[e]; a0[e + 8] = hi0[e];
          a1[e] = lo1[e]; a1[e + 8] = hi1[e];
        }
        accMel0 = __builtin_amdgcn_wmma_f32_16x16x32_f16(false, a0, false, bm, (short)0, accMel0, false, false);
        accMel1 = __builtin_amdgcn_wmma_f32_16x16x32_f16(false, a1, false, bm, (short)0, accMel1, false, false);
      }
    }
  }
  __syncthreads();

  // log + store: out[b][frame][mel][fi]
  if (wave < 5) {
    const int j = wave * 16 + lm;
    #pragma unroll
    for (int r = 0; r < 8; ++r) {
      const int n0 = fr0 + (r + lh * 8);
      const int n1 = n0 + 16;
      if (n0 < N0)
        out[(((size_t)b * N0 + n0) * NMEL + j) * 3 + fi] = logf(accMel0[r] + 1e-16f);
      if (n1 < N0)
        out[(((size_t)b * N0 + n1) * NMEL + j) * 3 + fi] = logf(accMel1[r] + 1e-16f);
    }
  }
}

extern "C" void kernel_launch(void* const* d_in, const int* in_sizes, int n_in,
                              void* d_out, int out_size, void* d_ws, size_t ws_size,
                              hipStream_t stream) {
  (void)in_sizes; (void)n_in; (void)out_size; (void)d_ws; (void)ws_size;
  const float* x     = (const float*)d_in[0];
  const float* w1024 = (const float*)d_in[1];
  const float* m1024 = (const float*)d_in[2];
  const float* w2048 = (const float*)d_in[3];
  const float* m2048 = (const float*)d_in[4];
  const float* w4096 = (const float*)d_in[5];
  const float* m4096 = (const float*)d_in[6];
  float* out = (float*)d_out;

  dim3 grid((N0 + 31) / 32, BATCH);   // 63 frame tiles x 8 batches
  spectro_wmma<1024><<<grid, 256, 0, stream>>>(x, w1024, m1024, out, 0);
  spectro_wmma<2048><<<grid, 256, 0, stream>>>(x, w2048, m2048, out, 1);
  spectro_wmma<4096><<<grid, 256, 0, stream>>>(x, w4096, m4096, out, 2);
}
